// ExtendedMptAttention_49684181680345
// MI455X (gfx1250) — matrix operand order repaired and checked by reading, not compiled, not run on hardware
//
#include <hip/hip_runtime.h>
#include <hip/hip_bf16.h>

// ---------------------------------------------------------------------------
// ExtendedMptAttention for MI455X (gfx1250): bf16 WMMA for all dense GEMMs,
// fused sim+topk retrieval, gather-based cache attention, fp32 accumulation.
// ---------------------------------------------------------------------------

typedef __attribute__((ext_vector_type(16))) __bf16 v16bf;
typedef __attribute__((ext_vector_type(8)))  float  v8f;
typedef unsigned short ush;

#define NEGF (-3.402823466e38f)

static constexpr int S      = 512;
static constexpr int H      = 2048;
static constexpr int NH     = 16;
static constexpr int D      = 128;
static constexpr int SC     = 8192;
static constexpr int TOPK   = 16;
static constexpr int QKVN   = 3 * H;        // 6144
static constexpr float SIM_THRESHOLD = 0.25f;

union V16 { uint4 q[2]; ush us[16]; v16bf v; };

__device__ __forceinline__ ush f2bf(float f) {
    unsigned u = __float_as_uint(f);
    unsigned r = u + 0x7FFFu + ((u >> 16) & 1u);   // round-to-nearest-even
    return (ush)(r >> 16);
}
__device__ __forceinline__ float bf2f(ush s) {
    return __uint_as_float(((unsigned)s) << 16);
}

__device__ __forceinline__ v8f wmma_bf16(v16bf a, v16bf b, v8f c) {
    return __builtin_amdgcn_wmma_f32_16x16x32_bf16(
        /*neg_a=*/false, a, /*neg_b=*/false, b,
        /*c_mod=*/(short)0, c, /*reuse_a=*/false, /*reuse_b=*/false);
}

// A operand (16x32 bf16, row-major source, row stride ld elements).
// lane L: row = L&15; half (L>>4): elems 0..15 = K hi*16+0..15
__device__ __forceinline__ v16bf load_a_tile(const ush* base, int ld, int row0,
                                             int k0, int lane) {
    int r = lane & 15, hi = lane >> 4;
    const ush* p = base + (size_t)(row0 + r) * ld + k0;
    V16 t;
    t.q[0] = *(const uint4*)(p + hi * 16);
    t.q[1] = *(const uint4*)(p + hi * 16 + 8);
    return t.v;
}

// B operand (32x16 bf16). Column n of B is a contiguous-K row in memory
// (B[k][n] = src[(col0+n)*ld + k]).  lane L: col = L&15; K = (L>>4)*16 + e.
__device__ __forceinline__ v16bf load_b_tile(const ush* base, int ld, int col0,
                                             int k0, int lane) {
    int c = lane & 15, ks = (lane >> 4) * 16;
    const ush* p = base + (size_t)(col0 + c) * ld + k0 + ks;
    V16 t;
    t.q[0] = *(const uint4*)(p);
    t.q[1] = *(const uint4*)(p + 8);
    return t.v;
}

// ---------------------------------------------------------------------------
// 1) elementwise f32 -> bf16
__global__ void k_f32_to_bf16(const float* __restrict__ x, ush* __restrict__ y,
                              int n) {
    int i = blockIdx.x * 256 + threadIdx.x;
    if (i < n) y[i] = f2bf(x[i]);
}

// ---------------------------------------------------------------------------
// 2) C[M,N] = A[M,K](bf16) * B[N,K](bf16)^T, fp32 out.  1 wave = 16x16 tile.
__global__ void k_gemm_bf16_nt(const ush* __restrict__ A, const ush* __restrict__ B,
                               float* __restrict__ C, int M, int N, int K) {
    int lane = threadIdx.x & 31;
    int w    = threadIdx.x >> 5;
    int n0   = (blockIdx.x * 4 + w) * 16;
    int m0   = blockIdx.y * 16;
    v8f acc = {};
    for (int k0 = 0; k0 < K; k0 += 32) {
        __builtin_prefetch(B + (size_t)(n0 + (lane & 15)) * K + k0 + 256, 0, 1);
        v16bf a = load_a_tile(A, K, m0, k0, lane);
        v16bf b = load_b_tile(B, K, n0, k0, lane);
        acc = wmma_bf16(a, b, acc);
    }
    int n = lane & 15, hi = lane >> 4;
#pragma unroll
    for (int r = 0; r < 8; ++r)
        C[(size_t)(m0 + r + 8 * hi) * N + n0 + n] = acc[r];
}

// ---------------------------------------------------------------------------
// 3) split qkv [S,6144] into per-head bf16 q/k/v [NH][S][D] + 1/||q||.
__global__ void k_split_heads(const float* __restrict__ qkv, ush* __restrict__ qb,
                              ush* __restrict__ kb, ush* __restrict__ vb,
                              float* __restrict__ qninv) {
    int gw   = blockIdx.x * 8 + (threadIdx.x >> 5);   // wave id: NH*S = 8192
    int lane = threadIdx.x & 31;
    int h = gw >> 9, s = gw & 511;
    const float* src = qkv + (size_t)s * QKVN + h * D;
    ush* qd = qb + ((size_t)h * S + s) * D;
    ush* kd = kb + ((size_t)h * S + s) * D;
    ush* vd = vb + ((size_t)h * S + s) * D;
    float ss = 0.f;
#pragma unroll
    for (int t = 0; t < 4; ++t) {
        int d = t * 32 + lane;
        float q = src[d];
        ss += q * q;
        qd[d] = f2bf(q);
        kd[d] = f2bf(src[H + d]);
        vd[d] = f2bf(src[2 * H + d]);
    }
#pragma unroll
    for (int m = 16; m; m >>= 1) ss += __shfl_xor(ss, m, 32);
    if (lane == 0) qninv[h * S + s] = rsqrtf(ss);
}

// ---------------------------------------------------------------------------
// 4) k_cache -> bf16 + 1/||row||.  One wave per cache row (NH*SC rows).
__global__ void k_conv_kcache(const float* __restrict__ kc, ush* __restrict__ kcb,
                              float* __restrict__ kcninv) {
    int gw   = blockIdx.x * 8 + (threadIdx.x >> 5);
    int lane = threadIdx.x & 31;
    const float* src = kc + (size_t)gw * D;
    ush* dst = kcb + (size_t)gw * D;
    float ss = 0.f;
#pragma unroll
    for (int t = 0; t < 4; ++t) {
        int d = t * 32 + lane;
        float x = src[d];
        ss += x * x;
        dst[d] = f2bf(x);
    }
#pragma unroll
    for (int m = 16; m; m >>= 1) ss += __shfl_xor(ss, m, 32);
    if (lane == 0) kcninv[gw] = rsqrtf(ss);
}

// ---------------------------------------------------------------------------
// 5) fused cosine-sim (WMMA) + per-row top-16 over the 8192-entry cache.
//    1 wave per (head, 16-query tile).
__global__ void k_sim_topk(const ush* __restrict__ qb, const ush* __restrict__ kcb,
                           const float* __restrict__ qninv,
                           const float* __restrict__ kcninv,
                           int* __restrict__ tidx, float* __restrict__ tval) {
    __shared__ float tile[16 * 17];
    int lane = threadIdx.x & 31;
    int h = blockIdx.x >> 5, qt = blockIdx.x & 31;
    int s0 = qt * 16;
    const ush* qbase = qb + (size_t)h * S * D;
    v16bf qa[4];
#pragma unroll
    for (int t = 0; t < 4; ++t) qa[t] = load_a_tile(qbase, D, s0, t * 32, lane);
    int hi = lane >> 4, n = lane & 15;
    float qn[8];
#pragma unroll
    for (int r = 0; r < 8; ++r) qn[r] = qninv[h * S + s0 + r + 8 * hi];

    float topv[16];
    int   topi[16];
#pragma unroll
    for (int i = 0; i < 16; ++i) { topv[i] = NEGF; topi[i] = 0; }
    float curMin = NEGF;
    int   curPos = 0;

    const ush*   kcb_h = kcb + (size_t)h * SC * D;
    const float* kcn_h = kcninv + h * SC;

    for (int c0 = 0; c0 < SC; c0 += 16) {
        v8f acc = {};
#pragma unroll
        for (int t = 0; t < 4; ++t)
            acc = wmma_bf16(qa[t], load_b_tile(kcb_h, D, c0, t * 32, lane), acc);
        float kn = kcn_h[c0 + n];
#pragma unroll
        for (int r = 0; r < 8; ++r)
            tile[(r + 8 * hi) * 17 + n] = acc[r] * qn[r] * kn;
        __syncthreads();
        if (lane < 16) {
#pragma unroll
            for (int j = 0; j < 16; ++j) {
                float c = tile[lane * 17 + j];
                if (c > curMin) {
                    int ci = c0 + j;
#pragma unroll
                    for (int i = 0; i < 16; ++i)
                        if (i == curPos) { topv[i] = c; topi[i] = ci; }
                    curMin = topv[0]; curPos = 0;
#pragma unroll
                    for (int i = 1; i < 16; ++i)
                        if (topv[i] < curMin) { curMin = topv[i]; curPos = i; }
                }
            }
        }
        __syncthreads();
    }
    if (lane < 16) {
        int base = (h * S + s0 + lane) * TOPK;
#pragma unroll
        for (int i = 0; i < 16; ++i) { tidx[base + i] = topi[i]; tval[base + i] = topv[i]; }
    }
}

// ---------------------------------------------------------------------------
// 6) attention: cache scores (gather) + causal local scores (WMMA) + softmax
//    + PV (WMMA local, gather cache).  1 wave per (head, 16-query tile).
__global__ void k_attention(const float* __restrict__ qkvf, const ush* __restrict__ qb,
                            const ush* __restrict__ kb, const ush* __restrict__ vb,
                            const float* __restrict__ kcache,
                            const float* __restrict__ vcache,
                            const float* __restrict__ pbias,
                            const int* __restrict__ tidx, const float* __restrict__ tval,
                            ush* __restrict__ ctxb) {
    __shared__ float sc[16 * 544];
    __shared__ ush   pbf[16 * 544];
    __shared__ int   idxT[256];
    __shared__ float valT[256];
    __shared__ float rowsum[16];

    int lane = threadIdx.x & 31;
    int h = blockIdx.x >> 5, qt = blockIdx.x & 31;
    int s0 = qt * 16;
    const float scale = 0.08838834764831845f;   // 1/sqrt(128)

    // load this tile's topk indices / sim values
    for (int p = lane; p < 256; p += 32) {
        int m = p >> 4, j = p & 15;
        int base = (h * S + s0 + m) * TOPK + j;
        idxT[p] = tidx[base];
        valT[p] = tval[base];
    }
    __syncthreads();

    // ---- cache scores: q_m . k_cache[idx[m][j]]  (8 (m,j) pairs per lane)
    const float* kc_h = kcache + (size_t)h * SC * D;
#pragma unroll
    for (int pp = 0; pp < 8; ++pp) {
        int p = lane * 8 + pp;
        int m = p >> 4, j = p & 15;
        const float* qrow = qkvf + (size_t)(s0 + m) * QKVN + h * D;
        const float* krow = kc_h + (size_t)idxT[p] * D;
        float dot = 0.f;
        for (int d = 0; d < D; d += 4) {
            float4 qv = *(const float4*)(qrow + d);
            float4 kv = *(const float4*)(krow + d);
            dot += qv.x * kv.x + qv.y * kv.y + qv.z * kv.z + qv.w * kv.w;
        }
        sc[m * 544 + j] = (valT[p] > SIM_THRESHOLD) ? dot * scale : NEGF;
    }

    // ---- local causal scores via WMMA + position bias
    const ush* qbase = qb + (size_t)h * S * D;
    const ush* kbase = kb + (size_t)h * S * D;
    v16bf qa[4];
#pragma unroll
    for (int t = 0; t < 4; ++t) qa[t] = load_a_tile(qbase, D, s0, t * 32, lane);
    int hi = lane >> 4, n = lane & 15;
    const float* pb_h = pbias + (size_t)h * S * S;
    for (int k0 = 0; k0 <= s0; k0 += 16) {
        v8f acc = {};
#pragma unroll
        for (int t = 0; t < 4; ++t)
            acc = wmma_bf16(qa[t], load_b_tile(kbase, D, k0, t * 32, lane), acc);
#pragma unroll
        for (int r = 0; r < 8; ++r) {
            int m = r + 8 * hi;
            float sv = acc[r] * scale + pb_h[(size_t)(s0 + m) * S + k0 + n];
            if (k0 + n > s0 + m) sv = NEGF;           // causal
            sc[m * 544 + 16 + k0 + n] = sv;
        }
    }
    __syncthreads();

    // ---- softmax per row (16 cache slots + s0+16 local keys)
    int L    = 16 + s0 + 16;
    int Lpad = 16 + ((s0 + 16 + 31) & ~31);
    if (lane < 16) {
        float* row = sc + lane * 544;
        float mx = NEGF;
        for (int c = 0; c < L; ++c) mx = fmaxf(mx, row[c]);
        float sum = 0.f;
        ush* prow = pbf + lane * 544;
        for (int c = 0; c < L; ++c) {
            float e = __expf(row[c] - mx);
            sum += e;
            prow[c] = f2bf(e);
        }
        for (int c = L; c < Lpad; ++c) prow[c] = 0;   // zero-pad for WMMA A tile
        rowsum[lane] = sum;
    }
    __syncthreads();

    // ---- PV: local part via WMMA (P bf16 from LDS x V bf16), 8 d-tiles
    v8f acc8[8] = {};
    const ush* vb_h = vb + (size_t)h * S * D;
    for (int kb2 = 0; kb2 < s0 + 16; kb2 += 32) {
        V16 ta;
        const ush* pr = pbf + (lane & 15) * 544 + 16 + kb2 + hi * 16;
        ta.q[0] = *(const uint4*)(pr);
        ta.q[1] = *(const uint4*)(pr + 8);
        v16bf a = ta.v;
#pragma unroll
        for (int nd = 0; nd < 8; ++nd) {
            V16 tb;
            int d = nd * 16 + n;
            const ush* vp = vb_h + (size_t)(kb2 + hi * 16) * D + d;
#pragma unroll
            for (int e = 0; e < 16; ++e) tb.us[e] = vp[(size_t)e * D];
            acc8[nd] = wmma_bf16(a, tb.v, acc8[nd]);
        }
    }

    // ---- PV: cache part (gathered v_cache rows, fp32 VALU)
    const float* vc_h = vcache + (size_t)h * SC * D;
#pragma unroll
    for (int r = 0; r < 8; ++r) {
        int m = r + 8 * hi;
        for (int j = 0; j < 16; ++j) {
            float w = bf2f(pbf[m * 544 + j]);
            if (w > 0.f) {
                const float* vrow = vc_h + (size_t)idxT[m * 16 + j] * D;
#pragma unroll
                for (int nd = 0; nd < 8; ++nd)
                    acc8[nd][r] += w * vrow[nd * 16 + n];
            }
        }
    }

    // ---- normalize + emit ctx (bf16, [S, H] layout for the output GEMM)
#pragma unroll
    for (int r = 0; r < 8; ++r) {
        int m = r + 8 * hi;
        float inv = 1.0f / rowsum[m];
#pragma unroll
        for (int nd = 0; nd < 8; ++nd)
            ctxb[(size_t)(s0 + m) * H + h * D + nd * 16 + n] = f2bf(acc8[nd][r] * inv);
    }
}

// ---------------------------------------------------------------------------
extern "C" void kernel_launch(void* const* d_in, const int* in_sizes, int n_in,
                              void* d_out, int out_size, void* d_ws, size_t ws_size,
                              hipStream_t stream) {
    (void)in_sizes; (void)n_in; (void)out_size; (void)ws_size;
    const float* hs   = (const float*)d_in[0];   // [1,512,2048]
    const float* Wqkv = (const float*)d_in[1];   // [6144,2048]
    const float* Wout = (const float*)d_in[2];   // [2048,2048]
    const float* kc   = (const float*)d_in[3];   // [1,16,8192,128]
    const float* vc   = (const float*)d_in[4];   // [1,16,8192,128]
    const float* pb   = (const float*)d_in[5];   // [16,512,512]
    // d_in[6] attention_mask == causal (hardcoded), d_in[7] topk == 16 (hardcoded)
    float* out = (float*)d_out;

    // workspace carve-up (256B aligned)
    char* w = (char*)d_ws;
    auto take = [&](size_t bytes) {
        char* p = w;
        w += (bytes + 255) & ~(size_t)255;
        return p;
    };
    ush*   hsb    = (ush*)  take((size_t)S * H * 2);
    ush*   wqkvb  = (ush*)  take((size_t)QKVN * H * 2);
    ush*   woutb  = (ush*)  take((size_t)H * H * 2);
    float* qkvf   = (float*)take((size_t)S * QKVN * 4);
    ush*   qb     = (ush*)  take((size_t)NH * S * D * 2);
    ush*   kb     = (ush*)  take((size_t)NH * S * D * 2);
    ush*   vb     = (ush*)  take((size_t)NH * S * D * 2);
    float* qninv  = (float*)take((size_t)NH * S * 4);
    ush*   kcb    = (ush*)  take((size_t)NH * SC * D * 2);
    float* kcninv = (float*)take((size_t)NH * SC * 4);
    int*   tidx   = (int*)  take((size_t)NH * S * TOPK * 4);
    float* tval   = (float*)take((size_t)NH * S * TOPK * 4);
    ush*   ctxb   = (ush*)  take((size_t)S * H * 2);

    // 1) f32 -> bf16 converts
    {
        int n1 = S * H;       // 1,048,576
        int n2 = QKVN * H;    // 12,582,912
        int n3 = H * H;       // 4,194,304
        k_f32_to_bf16<<<(n1 + 255) / 256, 256, 0, stream>>>(hs, hsb, n1);
        k_f32_to_bf16<<<(n2 + 255) / 256, 256, 0, stream>>>(Wqkv, wqkvb, n2);
        k_f32_to_bf16<<<(n3 + 255) / 256, 256, 0, stream>>>(Wout, woutb, n3);
    }
    // 2) QKV projection: [512,2048] x [6144,2048]^T -> [512,6144]
    k_gemm_bf16_nt<<<dim3(QKVN / 64, S / 16), 128, 0, stream>>>(hsb, wqkvb, qkvf,
                                                                S, QKVN, H);
    // 3) head split + q norms
    k_split_heads<<<(NH * S) / 8, 256, 0, stream>>>(qkvf, qb, kb, vb, qninv);
    // 4) k_cache -> bf16 + norms
    k_conv_kcache<<<(NH * SC) / 8, 256, 0, stream>>>(kc, kcb, kcninv);
    // 5) fused sim + top-16
    k_sim_topk<<<NH * (S / 16), 32, 0, stream>>>(qb, kcb, qninv, kcninv, tidx, tval);
    // 6) attention
    k_attention<<<NH * (S / 16), 32, 0, stream>>>(qkvf, qb, kb, vb, kc, vc, pb,
                                                  tidx, tval, ctxb);
    // 7) output projection: [512,2048] x [2048,2048]^T -> d_out
    k_gemm_bf16_nt<<<dim3(H / 64, S / 16), 128, 0, stream>>>(ctxb, woutb, out,
                                                             S, H, H);
}
